// ISTFT_62036507623564
// MI455X (gfx1250) — matrix operand order repaired and checked by reading, not compiled, not run on hardware
//
#include <hip/hip_runtime.h>
#include <hip/hip_bf16.h>

typedef __attribute__((ext_vector_type(16))) _Float16 v16h;
typedef __attribute__((ext_vector_type(8)))  _Float16 v8h;
typedef __attribute__((ext_vector_type(8)))  float    v8f;

// AS(3) pointer type: ptrtoint of this yields the raw LDS byte offset that
// GLOBAL_LOAD_ASYNC_TO_LDS_* wants in its VDST operand.
typedef __attribute__((address_space(3))) _Float16 lds_half;

namespace {
constexpr int kBatch  = 16;
constexpr int kCutoff = 513;
constexpr int kK      = 1026;    // 2*CUTOFF
constexpr int kKpad   = 1056;    // 33 * 32 (zero-padded K)
constexpr int kNfft   = 1024;
constexpr int kT      = 2048;
constexpr int kOutLen = 523008;  // (T-1)*256 + NFFT - 2*NFFT
constexpr int BM = 128, BN = 128, BKc = 32;
constexpr int kRow = 40;         // LDS row stride (halves): 80B = 16B-aligned,
                                 // 20-bank stride -> conflict-free b128 access
}

// ---------------------------------------------------------------------------
// Kernel 0: invbasis [K=1026][N=1024] f32  ->  invT [N=1024][Kpad=1056] f16
// Transposed + zero-padded so B tiles are contiguous 64B runs per row.
// ---------------------------------------------------------------------------
__global__ __launch_bounds__(256)
void istft_cvt_basis(const float* __restrict__ inv, _Float16* __restrict__ invT) {
    const int id = blockIdx.x * 256 + threadIdx.x;   // grid covers 1024*1056 exactly
    const int n  = id / kKpad;
    const int k  = id - n * kKpad;
    const float v = (k < kK) ? inv[(size_t)k * kNfft + n] : 0.0f;
    invT[id] = (_Float16)v;
}

// ---------------------------------------------------------------------------
// Kernel 1: frames[b,t,:] = spec[b,:,t] @ invbasis   (f16 WMMA, f32 accum)
//   spec built on the fly (mag*cos / mag*sin), selection made *scalar* via
//   readfirstlane (it is wave-uniform). B tile staged to LDS per K-step with
//   GLOBAL_LOAD_ASYNC_TO_LDS_B128 (ASYNCcnt), overlapping the trig VALU work.
//   Block = 256 thr (8 waves), tile 128x128, K-step 32; wave = 64x32 sub-tile.
// ---------------------------------------------------------------------------
__global__ __launch_bounds__(256)
void istft_gemm(const float* __restrict__ mag, const float* __restrict__ ang,
                const _Float16* __restrict__ invT, float* __restrict__ frames)
{
    __shared__ __align__(16) _Float16 sA[BM][kRow];   // 10 KB
    __shared__ __align__(16) _Float16 sB[BN][kRow];   // 10 KB

    const int b  = blockIdx.z;
    const int m0 = blockIdx.y * BM;          // frame (t) tile base
    const int n0 = blockIdx.x * BN;          // in-frame sample (k) tile base

    const int tid  = threadIdx.x;
    const int lane = tid & 31;
    const int wv   = tid >> 5;
    const int wm   = (wv >> 2) * 64;         // wave M offset inside tile
    const int wn   = (wv & 3)  * 32;         // wave N offset inside tile
    const int half = lane >> 4;              // lane group (ISA 16-bit layouts)
    const int l16  = lane & 15;

    v8f acc[4][2] = {};

    // A staging: thread stages column t = m0 + (tid&127), 16 K-values.
    // cGrp is constant across each wave32 -> make it an SGPR so the cos/sin/pad
    // selection compiles to scalar branches instead of exec masking.
    const int tLoc = tid & 127;
    const int cGrp = __builtin_amdgcn_readfirstlane((tid >> 7) << 4);
    const size_t base = (size_t)b * kCutoff * kT + (size_t)(m0 + tLoc);

    // B async staging: 128 rows x 64B = 512 x 16B chunks -> 2 chunks/thread.
    const int rB0 = tid >> 2;                // rows 0..63
    const int rB1 = 64 + (tid >> 2);         // rows 64..127
    const int chH = (tid & 3) * 8;           // 16B chunk offset within row (halves)
    const _Float16* gB = invT + (size_t)n0 * kKpad;

    for (int kc = 0; kc < kKpad; kc += BKc) {
        // ---- kick off async B-tile copy (runs under ASYNCcnt while we do trig)
        {
            unsigned long long g0 =
                (unsigned long long)(gB + (size_t)rB0 * kKpad + kc + chH);
            unsigned long long g1 =
                (unsigned long long)(gB + (size_t)rB1 * kKpad + kc + chH);
            unsigned l0 = (unsigned)(size_t)(lds_half*)&sB[rB0][chH];
            unsigned l1 = (unsigned)(size_t)(lds_half*)&sB[rB1][chH];
            asm volatile("global_load_async_to_lds_b128 %0, %1, off"
                         :: "v"(l0), "v"(g0) : "memory");
            asm volatile("global_load_async_to_lds_b128 %0, %1, off"
                         :: "v"(l1), "v"(g1) : "memory");
        }

        // ---- stage spec tile -> sA (coalesced across lanes: consecutive t) ----
        #pragma unroll
        for (int i = 0; i < 16; ++i) {
            const int cl = cGrp + i;
            const int c  = kc + cl;                      // wave-uniform (SGPR)
            float v = 0.0f;                              // K-pad rows -> exact 0
            if (c < kCutoff) {
                const size_t off = base + (size_t)c * kT;
                v = mag[off] * __cosf(ang[off]);
            } else if (c < kK) {
                const size_t off = base + (size_t)(c - kCutoff) * kT;
                v = mag[off] * __sinf(ang[off]);
            }
            sA[tLoc][cl] = (_Float16)v;
        }

        // drain the async B copy, then rendezvous
        asm volatile("s_wait_asynccnt 0x0" ::: "memory");
        __syncthreads();

        // ---- B fragments from LDS: lane = col N=l16, v0-3=K 8h.., v4-7=K 16+8h.. (2x b128)
        v16h bf[2];
        #pragma unroll
        for (int f = 0; f < 2; ++f) {
            const _Float16* p = &sB[wn + 16 * f + l16][8 * half];
            v8h lo = *(const v8h*)(p);
            v8h hi = *(const v8h*)(p + 16);
            bf[f] = __builtin_shufflevector(lo, hi,
                      0,1,2,3,4,5,6,7,8,9,10,11,12,13,14,15);
        }

        // ---- A fragments from LDS: lane row M=l16, v0-3=K 8h.., v4-7=K 16+8h..
        v16h af[4];
        #pragma unroll
        for (int i = 0; i < 4; ++i) {
            const _Float16* p = &sA[wm + 16 * i + l16][8 * half];
            v8h lo = *(const v8h*)(p);
            v8h hi = *(const v8h*)(p + 16);
            af[i] = __builtin_shufflevector(lo, hi,
                      0,1,2,3,4,5,6,7,8,9,10,11,12,13,14,15);
        }

        // ---- 8x v_wmma_f32_16x16x32_f16 per K-step ----
        #pragma unroll
        for (int i = 0; i < 4; ++i)
            #pragma unroll
            for (int f = 0; f < 2; ++f)
                acc[i][f] = __builtin_amdgcn_wmma_f32_16x16x32_f16(
                    false, af[i], false, bf[f], (short)0, acc[i][f], false, false);

        __syncthreads();
    }

    // ---- epilogue: C/D layout (N=l16, M=r+8*half per VGPR r) -> frames ----
    #pragma unroll
    for (int i = 0; i < 4; ++i) {
        const int t = m0 + wm + 16 * i;
        #pragma unroll
        for (int f = 0; f < 2; ++f) {
            const int k = n0 + wn + 16 * f + l16;
            float* dst = frames + ((size_t)b * kT + t) * kNfft + k;
            #pragma unroll
            for (int r = 0; r < 8; ++r)
                dst[(size_t)(r + 8 * half) * kNfft] = acc[i][f][r];
        }
    }
}

// ---------------------------------------------------------------------------
// Kernel 2: overlap-add gather. In the trimmed region every sample is the sum
// of exactly 4 frame taps: out[b][s] = sum_j frames[b][q-j][256j+p],
// q=(s+1024)>>8, p=(s+1024)&255. Deterministic, no atomics, no bounds checks.
// ---------------------------------------------------------------------------
__global__ __launch_bounds__(256)
void istft_ola(const float* __restrict__ frames, float* __restrict__ out) {
    const int b  = blockIdx.y;
    const int s  = blockIdx.x * 256 + threadIdx.x;   // kOutLen = 256*2043 exactly
    const int sf = s + kNfft;
    const int q  = sf >> 8;
    const int p  = sf & 255;
    const float* fb = frames + (size_t)b * kT * kNfft;
    float a = 0.0f;
    #pragma unroll
    for (int j = 0; j < 4; ++j)
        a += fb[(size_t)(q - j) * kNfft + (j << 8) + p];
    out[(size_t)b * kOutLen + s] = a;
}

// ---------------------------------------------------------------------------
// Workspace layout (needs ~136.4 MB):
//   [0, 2.16MB)        : invT  f16 [1024][1056]
//   [2.16MB, +134.2MB) : frames f32 [16][2048][1024]
// ---------------------------------------------------------------------------
extern "C" void kernel_launch(void* const* d_in, const int* in_sizes, int n_in,
                              void* d_out, int out_size, void* d_ws, size_t ws_size,
                              hipStream_t stream) {
    (void)in_sizes; (void)n_in; (void)out_size; (void)ws_size;

    const float* mag = (const float*)d_in[0];
    const float* ang = (const float*)d_in[1];
    const float* inv = (const float*)d_in[2];
    float*       out = (float*)d_out;

    _Float16* invT   = (_Float16*)d_ws;
    float*    frames = (float*)((char*)d_ws + (size_t)kNfft * kKpad * sizeof(_Float16));

    istft_cvt_basis<<<dim3((kNfft * kKpad) / 256), 256, 0, stream>>>(inv, invT);
    istft_gemm<<<dim3(kNfft / BN, kT / BM, kBatch), 256, 0, stream>>>(mag, ang, invT, frames);
    istft_ola<<<dim3(kOutLen / 256, kBatch), 256, 0, stream>>>(frames, out);
}
